// Block_47940424958222
// MI455X (gfx1250) — matrix operand order repaired and checked by reading, not compiled, not run on hardware
//
#include <hip/hip_runtime.h>
#include <math.h>
#include <stdint.h>

// ---------------------------------------------------------------------------
// Types for CDNA5 WMMA (wave32): V_WMMA_F32_16X16X32_BF16
// ---------------------------------------------------------------------------
typedef __attribute__((ext_vector_type(16))) __bf16 bf16x16;
typedef __attribute__((ext_vector_type(8)))  __bf16 bf16x8;
typedef __attribute__((ext_vector_type(8)))  float  f32x8;

union BFrag  { bf16x16 v; bf16x8 h[2]; __bf16 e[16]; };
union BHalf8 { bf16x8 v; __bf16 e[8]; };

static __device__ inline bf16x16 make_frag(const __bf16* p0, const __bf16* p1) {
  BFrag u;
  u.h[0] = *(const bf16x8*)p0;   // 16B aligned chunk
  u.h[1] = *(const bf16x8*)p1;   // 16B aligned chunk
  return u.v;
}

static __device__ inline f32x8 wmma_bf16(bf16x16 a, bf16x16 b, f32x8 c) {
  return __builtin_amdgcn_wmma_f32_16x16x32_bf16(false, a, false, b, (short)0, c,
                                                 false, false);
}

// Async global->LDS copy (CDNA5 TDM-adjacent path, tracked by ASYNCcnt).
// lds_off: LDS byte address (low 32 bits of a generic LDS pointer);
// gptr: 64-bit global address (VGPR pair).
static __device__ inline void async_load_b128(uint32_t lds_off, const void* gptr) {
  asm volatile("global_load_async_to_lds_b128 %0, %1, off"
               :: "v"(lds_off), "v"((uint64_t)(uintptr_t)gptr)
               : "memory");
}
static __device__ inline void wait_async() {
  asm volatile("s_wait_asynccnt 0x0" ::: "memory");
}
static __device__ inline uint32_t lds_addr(const void* p) {
  return (uint32_t)(uintptr_t)p;  // LDS aperture: addr[31:0] = LDS byte offset
}

// 16-lane butterfly reductions via v_permlane16_b32 (VALU, not LDS).
static __device__ inline float permx16(float v, unsigned lo, unsigned hi) {
  int i = __float_as_int(v);
  i = __builtin_amdgcn_permlane16(i, i, (int)lo, (int)hi, false, false);
  return __int_as_float(i);
}
static __device__ inline float redmax16(float x) {
  x = fmaxf(x, permx16(x, 0x67452301u, 0xEFCDAB89u));  // xor 1
  x = fmaxf(x, permx16(x, 0x54761032u, 0xDCFE98BAu));  // xor 2
  x = fmaxf(x, permx16(x, 0x32107654u, 0xBA98FEDCu));  // xor 4
  x = fmaxf(x, permx16(x, 0xFEDCBA98u, 0x76543210u));  // xor 8
  return x;
}
static __device__ inline float redsum16(float x) {
  x += permx16(x, 0x67452301u, 0xEFCDAB89u);
  x += permx16(x, 0x54761032u, 0xDCFE98BAu);
  x += permx16(x, 0x32107654u, 0xBA98FEDCu);
  x += permx16(x, 0xFEDCBA98u, 0x76543210u);
  return x;
}

// Problem constants
#define SEQ    2048
#define BATCH  2
#define CDIM   1024
#define HEADS  16
#define DH     64
#define HID    4096
#define MTOK   (BATCH * SEQ)     // 4096 activation rows

// ---------------------------------------------------------------------------
// Weight prep: fp32 [K][N] -> bf16 [N][K] (transposed through LDS tile)
// ---------------------------------------------------------------------------
__global__ __launch_bounds__(256) void cvt_transpose(const float* __restrict__ in,
                                                     __bf16* __restrict__ out,
                                                     int K, int N) {
  __shared__ float t[32][33];
  const int k0 = blockIdx.y * 32, n0 = blockIdx.x * 32;
  const int tx = threadIdx.x & 31, ty = threadIdx.x >> 5;
#pragma unroll
  for (int i = 0; i < 4; ++i)
    t[ty + i * 8][tx] = in[(size_t)(k0 + ty + i * 8) * N + n0 + tx];
  __syncthreads();
#pragma unroll
  for (int i = 0; i < 4; ++i)
    out[(size_t)(n0 + ty + i * 8) * K + k0 + tx] = (__bf16)t[tx][ty + i * 8];
}

// ---------------------------------------------------------------------------
// LayerNorm over last dim (1024), fp32 in -> bf16 out
// ---------------------------------------------------------------------------
__global__ __launch_bounds__(256) void ln_kernel(const float* __restrict__ x,
                                                 const float* __restrict__ g,
                                                 const float* __restrict__ b,
                                                 __bf16* __restrict__ out) {
  __shared__ float red[256];
  const int row = blockIdx.x;
  const int tid = threadIdx.x;
  const float* xr = x + (size_t)row * CDIM;
  float vals[4];
  float s = 0.f, ss = 0.f;
#pragma unroll
  for (int i = 0; i < 4; ++i) {
    float v = xr[tid + i * 256];
    vals[i] = v; s += v; ss += v * v;
  }
  red[tid] = s; __syncthreads();
  for (int st = 128; st > 0; st >>= 1) { if (tid < st) red[tid] += red[tid + st]; __syncthreads(); }
  float mu = red[0] * (1.f / CDIM);
  __syncthreads();
  red[tid] = ss; __syncthreads();
  for (int st = 128; st > 0; st >>= 1) { if (tid < st) red[tid] += red[tid + st]; __syncthreads(); }
  float var = red[0] * (1.f / CDIM) - mu * mu;
  float rstd = rsqrtf(var + 1e-5f);
#pragma unroll
  for (int i = 0; i < 4; ++i) {
    int c = tid + i * 256;
    out[(size_t)row * CDIM + c] = (__bf16)((vals[i] - mu) * rstd * g[c] + b[c]);
  }
}

// ---------------------------------------------------------------------------
// bf16 WMMA GEMM: C[M,N] = A[M,K] @ W[K,N] with W pre-transposed as Wt[N][K].
// Block tile 256x128, 256 threads (8 waves), wave tile 64x64 (4x4 WMMA,
// 0.5 frag-loads per WMMA), K step 32. Double-buffered LDS fed by
// global_load_async_to_lds_b128; one barrier + one asynccnt drain per K step.
// MODE 0: out bf16 = acc
// MODE 1: out f32  = acc + bias + resid         (proj, fc2)
// MODE 2: out bf16 = gelu_exact(acc + bias)     (fc1)
// MODE 3: qkv: Q/K cols -> bf16 row-major; V cols -> vT[col][row] (b128 stores)
// ---------------------------------------------------------------------------
#define LDT 40              // padded halfs per staged row
#define ASZ (256 * LDT)     // A tile: 256 rows x 32 halfs
#define BSZ (128 * LDT)     // B tile: 128 rows x 32 halfs

template <int MODE>
__global__ __launch_bounds__(256) void gemm_bf16_kernel(
    const __bf16* __restrict__ A, const __bf16* __restrict__ Wt,
    int M, int N, int K,
    const float* __restrict__ bias, const float* __restrict__ resid,
    float* __restrict__ outF, __bf16* __restrict__ outB,
    __bf16* __restrict__ outVT) {
  __shared__ __attribute__((aligned(16))) __bf16 sA[2 * ASZ];
  __shared__ __attribute__((aligned(16))) __bf16 sB[2 * BSZ];  // Wt rows: [n][k]

  const int tid = threadIdx.x;
  const int lane = tid & 31;
  const int w = tid >> 5;
  const int bm = blockIdx.y * 256;
  const int bn = blockIdx.x * 128;
  const int wm = (w >> 1) * 64;
  const int wn = (w & 1) * 64;

  // Staging maps: A one row per thread (256 rows x 32 halfs), B two threads/row
  const __bf16* gA = A + (size_t)(bm + tid) * K;
  const int brow = tid >> 1;
  const int bcol = (tid & 1) * 16;
  const __bf16* gB = Wt + (size_t)(bn + brow) * K + bcol;

  f32x8 acc[4][4];
#pragma unroll
  for (int mi = 0; mi < 4; ++mi)
#pragma unroll
    for (int nj = 0; nj < 4; ++nj) acc[mi][nj] = (f32x8)0.0f;

  auto stage = [&](int buf, int k0) {
    uint32_t da = lds_addr(sA + buf * ASZ + tid * LDT);
#pragma unroll
    for (int i = 0; i < 4; ++i) async_load_b128(da + 16 * i, gA + k0 + 8 * i);
    uint32_t db = lds_addr(sB + buf * BSZ + brow * LDT + bcol);
    async_load_b128(db, gB + k0);
    async_load_b128(db + 16, gB + k0 + 8);
  };

  stage(0, 0);
  wait_async();
  __syncthreads();

  for (int k0 = 0; k0 < K; k0 += 32) {
    const int cur = (k0 >> 5) & 1;
    const bool more = (k0 + 32) < K;
    if (more) stage(cur ^ 1, k0 + 32);  // async into the other buffer

    const __bf16* bA = sA + cur * ASZ;
    const __bf16* bB = sB + cur * BSZ;
    bf16x16 af[4], bfg[4];
#pragma unroll
    for (int mi = 0; mi < 4; ++mi) {
      int r = wm + mi * 16 + (lane & 15);
      int kb = (lane >> 4) * 16;
      af[mi] = make_frag(bA + r * LDT + kb, bA + r * LDT + kb + 8);
    }
#pragma unroll
    for (int nj = 0; nj < 4; ++nj) {
      int n = wn + nj * 16 + (lane & 15);
      int kb2 = (lane >> 4) * 16;
      bfg[nj] = make_frag(bB + n * LDT + kb2, bB + n * LDT + kb2 + 8);
    }
#pragma unroll
    for (int mi = 0; mi < 4; ++mi)
#pragma unroll
      for (int nj = 0; nj < 4; ++nj)
        acc[mi][nj] = wmma_bf16(af[mi], bfg[nj], acc[mi][nj]);

    wait_async();
    __syncthreads();
  }

  // Epilogue: VGPR r holds row (r + 8*(lane>=16)), col = lane&15
  const int rbase = (lane >> 4) * 8;
  const int cl = lane & 15;
  if (MODE == 3 && bn >= 2 * CDIM) {
    // V third: store transposed vT[col][row]; 8 consecutive rows per lane ->
    // one contiguous b128 store per tile.
#pragma unroll
    for (int mi = 0; mi < 4; ++mi)
#pragma unroll
      for (int nj = 0; nj < 4; ++nj) {
        int vcol = bn + wn + nj * 16 + cl - 2 * CDIM;
        int row0 = bm + wm + mi * 16 + rbase;
        BHalf8 p;
#pragma unroll
        for (int r = 0; r < 8; ++r) p.e[r] = (__bf16)acc[mi][nj][r];
        *(bf16x8*)(outVT + (size_t)vcol * M + row0) = p.v;
      }
    return;
  }
#pragma unroll
  for (int mi = 0; mi < 4; ++mi)
#pragma unroll
    for (int nj = 0; nj < 4; ++nj)
#pragma unroll
      for (int r = 0; r < 8; ++r) {
        int row = bm + wm + mi * 16 + rbase + r;
        int col = bn + wn + nj * 16 + cl;
        size_t idx = (size_t)row * N + col;
        float v = acc[mi][nj][r];
        if (MODE == 1 || MODE == 2) v += bias[col];
        if (MODE == 1) {
          outF[idx] = v + resid[idx];
        } else if (MODE == 2) {
          outB[idx] = (__bf16)(0.5f * v * (1.0f + erff(v * 0.70710678118654752f)));
        } else {
          outB[idx] = (__bf16)v;
        }
      }
}

// ---------------------------------------------------------------------------
// Flash attention: block = 64 q rows of one (batch, head); 4 waves, each owns
// 16 q rows. 32 keys per iteration; K/Vt double-buffered via async-to-LDS
// (one barrier + asynccnt drain per iteration). V comes pre-transposed from
// the QKV GEMM (vT[dh][token]).
// ---------------------------------------------------------------------------
#define KSZ (32 * 72)
#define VSZ (64 * 40)

__global__ __launch_bounds__(128) void attn_kernel(const __bf16* __restrict__ qkv,
                                                   const __bf16* __restrict__ vT,
                                                   __bf16* __restrict__ o) {
  __shared__ __attribute__((aligned(16))) __bf16 sQ[64 * 72];
  __shared__ __attribute__((aligned(16))) __bf16 sK[2 * KSZ];
  __shared__ __attribute__((aligned(16))) __bf16 sVt[2 * VSZ];    // [dh][key]
  __shared__ __attribute__((aligned(16))) __bf16 sP[4 * 16 * 40]; // per-wave P

  const int tid = threadIdx.x;
  const int lane = tid & 31;
  const int w = tid >> 5;
  const int qBase = blockIdx.x * 64;
  const int hIdx = blockIdx.y;
  const int b = blockIdx.z;
  const size_t tok0 = (size_t)b * SEQ;
  const int qoff = hIdx * DH;
  const int koff = CDIM + hIdx * DH;

  // Staging maps
  const int krow = tid >> 2;            // key row 0..31
  const int kc0 = (tid & 3) * 16;       // dh offset
  const int vrow = tid >> 1;            // dh row 0..63
  const int vc0 = (tid & 1) * 16;       // key offset
  const __bf16* gK = qkv + (tok0 + krow) * (3 * CDIM) + koff + kc0;
  const __bf16* gV = vT + (size_t)(hIdx * DH + vrow) * MTOK + tok0 + vc0;

  auto stageKV = [&](int buf, int j) {
    uint32_t dk = lds_addr(sK + buf * KSZ + krow * 72 + kc0);
    async_load_b128(dk, gK + (size_t)j * (3 * CDIM));
    async_load_b128(dk + 16, gK + (size_t)j * (3 * CDIM) + 8);
    uint32_t dv = lds_addr(sVt + buf * VSZ + vrow * 40 + vc0);
    async_load_b128(dv, gV + j);
    async_load_b128(dv + 16, gV + j + 8);
  };

  stageKV(0, 0);

  // Q tile 64x64 row-major (stride 72) — regular loads, single use
  {
    int row = tid >> 1;
    int c0 = (tid & 1) * 32;
    const __bf16* gq = qkv + (tok0 + qBase + row) * (3 * CDIM) + qoff + c0;
#pragma unroll
    for (int i = 0; i < 4; ++i)
      *(bf16x8*)(sQ + row * 72 + c0 + i * 8) = *(const bf16x8*)(gq + i * 8);
  }

  f32x8 oacc[4];
#pragma unroll
  for (int c = 0; c < 4; ++c) oacc[c] = (f32x8)0.0f;
  float mrow[8], lrow[8];
#pragma unroll
  for (int r = 0; r < 8; ++r) { mrow[r] = -1e30f; lrow[r] = 0.f; }

  wait_async();
  __syncthreads();

  for (int j = 0; j < SEQ; j += 32) {
    const int cur = (j >> 5) & 1;
    const bool more = (j + 32) < SEQ;
    if (more) stageKV(cur ^ 1, j + 32);  // async into the other buffer

    // S = Q @ K^T over Dh=64 (two K=32 panels), two 16-key tiles
    const __bf16* bK = sK + cur * KSZ;
    f32x8 s0 = (f32x8)0.0f, s1 = (f32x8)0.0f;
#pragma unroll
    for (int kp = 0; kp < 64; kp += 32) {
      int r = w * 16 + (lane & 15);
      int kb = (lane >> 4) * 16;
      bf16x16 aq = make_frag(sQ + r * 72 + kp + kb, sQ + r * 72 + kp + kb + 8);
      int n0 = lane & 15;
      int kb2 = (lane >> 4) * 16;
      bf16x16 b0 = make_frag(bK + n0 * 72 + kp + kb2, bK + n0 * 72 + kp + kb2 + 8);
      bf16x16 b1 = make_frag(bK + (n0 + 16) * 72 + kp + kb2,
                             bK + (n0 + 16) * 72 + kp + kb2 + 8);
      s0 = wmma_bf16(aq, b0, s0);
      s1 = wmma_bf16(aq, b1, s1);
    }

    // Online softmax (16-lane permlane reductions; row r of this wave's 16)
    const float scale = 0.125f;  // Dh^-0.5
#pragma unroll
    for (int r = 0; r < 8; ++r) {
      float v0 = s0[r] * scale;
      float v1 = s1[r] * scale;
      float mx = redmax16(fmaxf(v0, v1));
      float mnew = fmaxf(mrow[r], mx);
      float corr = __expf(mrow[r] - mnew);
      float p0 = __expf(v0 - mnew);
      float p1 = __expf(v1 - mnew);
      float rs = redsum16(p0 + p1);
      lrow[r] = lrow[r] * corr + rs;
      mrow[r] = mnew;
#pragma unroll
      for (int c = 0; c < 4; ++c) oacc[c][r] *= corr;
      int prow = (lane >> 4) * 8 + r;
      sP[w * 640 + prow * 40 + (lane & 15)] = (__bf16)p0;
      sP[w * 640 + prow * 40 + 16 + (lane & 15)] = (__bf16)p1;
    }
    // P write -> read is same-wave; per-wave DS ordering makes it safe.

    // O += P(16x32) @ V(32x64): 4 dh chunks
    {
      const __bf16* bV = sVt + cur * VSZ;
      int r = lane & 15;
      int kb = (lane >> 4) * 16;
      bf16x16 ap = make_frag(sP + w * 640 + r * 40 + kb,
                             sP + w * 640 + r * 40 + kb + 8);
#pragma unroll
      for (int c = 0; c < 4; ++c) {
        int n = c * 16 + (lane & 15);
        int kb2 = (lane >> 4) * 16;
        bf16x16 bv = make_frag(bV + n * 40 + kb2, bV + n * 40 + kb2 + 8);
        oacc[c] = wmma_bf16(ap, bv, oacc[c]);
      }
    }

    wait_async();
    __syncthreads();
  }

  // Normalize and write O bf16 into [B,N,C]
#pragma unroll
  for (int r = 0; r < 8; ++r) {
    float inv = 1.0f / lrow[r];
    int row = qBase + w * 16 + (lane >> 4) * 8 + r;
    size_t tok = tok0 + row;
#pragma unroll
    for (int c = 0; c < 4; ++c) {
      int col = hIdx * DH + c * 16 + (lane & 15);
      o[tok * CDIM + col] = (__bf16)(oacc[c][r] * inv);
    }
  }
}

// ---------------------------------------------------------------------------
// Launch
// ---------------------------------------------------------------------------
extern "C" void kernel_launch(void* const* d_in, const int* in_sizes, int n_in,
                              void* d_out, int out_size, void* d_ws, size_t ws_size,
                              hipStream_t stream) {
  const float* x      = (const float*)d_in[0];
  const float* ln1_g  = (const float*)d_in[1];
  const float* ln1_b  = (const float*)d_in[2];
  const float* qkv_w  = (const float*)d_in[3];
  const float* proj_w = (const float*)d_in[4];
  const float* proj_b = (const float*)d_in[5];
  const float* ln2_g  = (const float*)d_in[6];
  const float* ln2_b  = (const float*)d_in[7];
  const float* fc1_w  = (const float*)d_in[8];
  const float* fc1_b  = (const float*)d_in[9];
  const float* fc2_w  = (const float*)d_in[10];
  const float* fc2_b  = (const float*)d_in[11];
  float* out = (float*)d_out;

  char* ws = (char*)d_ws;
  size_t off = 0;
  auto carve = [&](size_t bytes) -> void* {
    void* p = ws + off;
    off += (bytes + 255) & ~(size_t)255;
    return p;
  };
  __bf16* WqkvT  = (__bf16*)carve((size_t)CDIM * 3 * CDIM * 2);  // [3C][C]
  __bf16* WprojT = (__bf16*)carve((size_t)CDIM * CDIM * 2);      // [C][C]
  __bf16* Wfc1T  = (__bf16*)carve((size_t)CDIM * HID * 2);       // [HID][C]
  __bf16* Wfc2T  = (__bf16*)carve((size_t)HID * CDIM * 2);       // [C][HID]
  __bf16* h1     = (__bf16*)carve((size_t)MTOK * CDIM * 2);
  __bf16* qkvB   = (__bf16*)carve((size_t)MTOK * 3 * CDIM * 2);  // V third unused
  __bf16* vT     = (__bf16*)carve((size_t)CDIM * MTOK * 2);      // [v_col][token]
  __bf16* oB     = (__bf16*)carve((size_t)MTOK * CDIM * 2);
  float*  x2     = (float*)carve((size_t)MTOK * CDIM * 4);
  __bf16* h2     = (__bf16*)carve((size_t)MTOK * CDIM * 2);
  __bf16* mlp    = (__bf16*)carve((size_t)MTOK * HID * 2);

  // Weight conversions (transposed to [N][K])
  cvt_transpose<<<dim3(3 * CDIM / 32, CDIM / 32), 256, 0, stream>>>(qkv_w, WqkvT, CDIM, 3 * CDIM);
  cvt_transpose<<<dim3(CDIM / 32, CDIM / 32), 256, 0, stream>>>(proj_w, WprojT, CDIM, CDIM);
  cvt_transpose<<<dim3(HID / 32, CDIM / 32), 256, 0, stream>>>(fc1_w, Wfc1T, CDIM, HID);
  cvt_transpose<<<dim3(CDIM / 32, HID / 32), 256, 0, stream>>>(fc2_w, Wfc2T, HID, CDIM);

  // LN1
  ln_kernel<<<MTOK, 256, 0, stream>>>(x, ln1_g, ln1_b, h1);

  // QKV = h1 @ Wqkv (Q/K row-major bf16; V written transposed into vT)
  gemm_bf16_kernel<3><<<dim3(3 * CDIM / 128, MTOK / 256), 256, 0, stream>>>(
      h1, WqkvT, MTOK, 3 * CDIM, CDIM, nullptr, nullptr, nullptr, qkvB, vT);

  // Attention
  attn_kernel<<<dim3(SEQ / 64, HEADS, BATCH), 128, 0, stream>>>(qkvB, vT, oB);

  // x2 = x + (o @ Wproj + proj_b)
  gemm_bf16_kernel<1><<<dim3(CDIM / 128, MTOK / 256), 256, 0, stream>>>(
      oB, WprojT, MTOK, CDIM, CDIM, proj_b, x, x2, nullptr, nullptr);

  // LN2
  ln_kernel<<<MTOK, 256, 0, stream>>>(x2, ln2_g, ln2_b, h2);

  // mlp = gelu(h2 @ Wfc1 + fc1_b)
  gemm_bf16_kernel<2><<<dim3(HID / 128, MTOK / 256), 256, 0, stream>>>(
      h2, Wfc1T, MTOK, HID, CDIM, fc1_b, nullptr, nullptr, mlp, nullptr);

  // out = x2 + (mlp @ Wfc2 + fc2_b)
  gemm_bf16_kernel<1><<<dim3(CDIM / 128, MTOK / 256), 256, 0, stream>>>(
      mlp, Wfc2T, MTOK, CDIM, HID, fc2_b, x2, out, nullptr, nullptr);
}
